// RelativeGlobalAttention_77360950936315
// MI455X (gfx1250) — matrix operand order repaired and checked by reading, not compiled, not run on hardware
//
#include <hip/hip_runtime.h>
#include <hip/hip_bf16.h>

// Relative Global Attention (Music-Transformer skew) for MI455X / gfx1250.
// All GEMMs on v_wmma_f32_16x16x32_f16 (f16 in, f32 accumulate).
// Srel[i,j] = q_i . Er[S-1+j-i]  (derivation of the _skew pad/reshape trick).

typedef __attribute__((ext_vector_type(16))) _Float16 v16h;
typedef __attribute__((ext_vector_type(8)))  _Float16 v8h;
typedef __attribute__((ext_vector_type(4)))  _Float16 v4h;
typedef __attribute__((ext_vector_type(8)))  float    v8f;
typedef int v4i_g __attribute__((vector_size(16)));   // matches builtin prototype

#define S_LEN   2048
#define DH      64
#define NH      16
#define NB      2
#define DM      1024
#define MAXLEN  2048

// ---- CDNA5 async global->LDS (ASYNCcnt path) ----
#if defined(__has_builtin)
#if __has_builtin(__builtin_amdgcn_global_load_async_to_lds_b128)
#define USE_ASYNC_LDS 1
#endif
#endif

__device__ __forceinline__ void cp_b128_to_lds(const void* g, void* l) {
#if USE_ASYNC_LDS
  __builtin_amdgcn_global_load_async_to_lds_b128(
      (__attribute__((address_space(1))) v4i_g*)g,
      (__attribute__((address_space(3))) v4i_g*)l, 0, 0);
#else
  *(uint4*)l = *(const uint4*)g;
#endif
}
__device__ __forceinline__ void async_wait_all() {
#if USE_ASYNC_LDS
#if __has_builtin(__builtin_amdgcn_s_wait_asynccnt)
  __builtin_amdgcn_s_wait_asynccnt(0);
#else
  asm volatile("s_wait_asynccnt 0x0" ::: "memory");
#endif
#endif
}

// ---------- WMMA fragment loaders (per CDNA5 ISA 7.12.2 layouts) ----------

// A-matrix 16x32 f16: lanes 0-15: M=lane, elems 0..7 = K 0..7, elems 8..15 = K 16..23
//                     lanes 16-31: M=lane-16, K offsets +8.
__device__ __forceinline__ v16h frag_a(const _Float16* p0, int row0, int stride) {
  const int lane = threadIdx.x & 31;
  const int l    = lane & 15;
  const int kb   = (lane >> 4) << 3;           // 0 or 8
  const _Float16* p = p0 + (size_t)(row0 + l) * stride + kb;
  v16h r;
#pragma unroll
  for (int t = 0; t < 8; ++t) { r[t] = p[t]; r[8 + t] = p[16 + t]; }
  return r;
}

// B-matrix 32x16 f16 stored [N][K] in LDS: lanes 0-15: N=lane, elems 0..7 = K 0..7, elems 8..15 = K 16..23;
//                                          lanes 16-31: N=lane-16, K offsets +8.
__device__ __forceinline__ v16h frag_b(const _Float16* p0, int row0, int stride) {
  const int lane = threadIdx.x & 31;
  const int l    = lane & 15;
  const int kb   = (lane >> 4) << 3;           // 0 or 8
  const _Float16* p = p0 + (size_t)(row0 + l) * stride + kb;
  v16h r;
#pragma unroll
  for (int t = 0; t < 8; ++t) { r[t] = p[t]; r[8 + t] = p[16 + t]; }
  return r;
}

__device__ __forceinline__ v8f vzero8() {
  v8f z;
#pragma unroll
  for (int t = 0; t < 8; ++t) z[t] = 0.0f;
  return z;
}

__device__ __forceinline__ v4h cvt4(float4 f) {
  v4h h;
  h[0] = (_Float16)f.x; h[1] = (_Float16)f.y;
  h[2] = (_Float16)f.z; h[3] = (_Float16)f.w;
  return h;
}

// row reductions within 16-lane halves (C-layout stripes N across 16 lanes)
__device__ __forceinline__ float red_max16(float v) {
#pragma unroll
  for (int m = 1; m <= 8; m <<= 1) v = fmaxf(v, __shfl_xor(v, m, 32));
  return v;
}
__device__ __forceinline__ float red_sum16(float v) {
#pragma unroll
  for (int m = 1; m <= 8; m <<= 1) v += __shfl_xor(v, m, 32);
  return v;
}

// ---------------- Projection: dst[b*H+h, s, d] = (X @ W^T + b) in f16 ----------------
// X: [B*S, DM] f32 ; W: [DM(out e), DM(in d)] f32 ; bias: [DM]
// Double-buffered LDS tiles, one barrier per K-step, next slice prefetched to regs.

__device__ __forceinline__ void stage_ld(const float* __restrict__ A, int row0, int k0,
                                         float4 v[4], int tid) {
#pragma unroll
  for (int t = 0; t < 4; ++t) {
    const int idx = tid + 128 * t;           // 512 float4 chunks of a 64x32 tile
    const int r = idx >> 3, c = idx & 7;
    v[t] = *(const float4*)&A[(size_t)(row0 + r) * DM + k0 + 4 * c];
  }
}
__device__ __forceinline__ void stage_st(_Float16* lds, const float4 v[4], int tid) {
#pragma unroll
  for (int t = 0; t < 4; ++t) {
    const int idx = tid + 128 * t;
    *(v4h*)&lds[idx * 4] = cvt4(v[t]);
  }
}

__global__ __launch_bounds__(128)
void proj_kernel(const float* __restrict__ X, const float* __restrict__ W,
                 const float* __restrict__ bias, _Float16* __restrict__ dst) {
  __shared__ __align__(16) _Float16 x_lds[2][64 * 32];
  __shared__ __align__(16) _Float16 w_lds[2][64 * 32];

  const int tid  = threadIdx.x;
  const int wv   = tid >> 5;          // wave id 0..3, owns rows 16*wv..16*wv+15
  const int lane = tid & 31;
  const int nl   = lane & 15;
  const int grp  = (lane >> 4) * 8;
  const int m0   = blockIdx.x * 64;   // rows (b*S+s)
  const int e0   = blockIdx.y * 64;   // output features

  v8f acc[4];
#pragma unroll
  for (int c = 0; c < 4; ++c) acc[c] = vzero8();

  float4 xr[4], wr[4];
  stage_ld(X, m0, 0, xr, tid);
  stage_ld(W, e0, 0, wr, tid);

  for (int kt = 0; kt < DM / 32; ++kt) {
    _Float16* xb = x_lds[kt & 1];
    _Float16* wb = w_lds[kt & 1];
    stage_st(xb, xr, tid);
    stage_st(wb, wr, tid);
    if (kt + 1 < DM / 32) {
      stage_ld(X, m0, (kt + 1) * 32, xr, tid);
      stage_ld(W, e0, (kt + 1) * 32, wr, tid);
    }
    __syncthreads();                   // single barrier: double buffering
    const v16h a = frag_a(xb, 16 * wv, 32);
#pragma unroll
    for (int c = 0; c < 4; ++c) {
      acc[c] = __builtin_amdgcn_wmma_f32_16x16x32_f16(
          false, a, false, frag_b(wb, c * 16, 32), (short)0, acc[c], false, false);
    }
  }

#pragma unroll
  for (int c = 0; c < 4; ++c) {
    const int e  = e0 + c * 16 + nl;
    const float bv = bias[e];
    const int h = e >> 6, d = e & 63;
#pragma unroll
    for (int r = 0; r < 8; ++r) {
      const int m = m0 + 16 * wv + r + grp;
      const int b = m >> 11;                 // S == 2048
      const int s = m & (S_LEN - 1);
      dst[(((size_t)(b * NH + h) * S_LEN) + s) * DH + d] = (_Float16)(acc[c][r] + bv);
    }
  }
}

// ---------------- Fused causal attention with relative positions ----------------
// One block (4 wave32) per 64 query rows per (b,h).
__global__ __launch_bounds__(128)
void attn_kernel(const _Float16* __restrict__ q_ws, const _Float16* __restrict__ k_ws,
                 const _Float16* __restrict__ v_ws, const float* __restrict__ Er,
                 float* __restrict__ out) {
  __shared__ __align__(16) _Float16 k_lds [64 * 64];   // [j][d]
  __shared__ __align__(16) _Float16 vt_lds[64 * 64];   // [d][j]  (V transposed)
  __shared__ __align__(16) _Float16 e_lds [128 * 64];  // [r][d]  rel-pos rows
  __shared__ __align__(16) _Float16 p_lds [64 * 128];  // staged Q*E^T
  __shared__ __align__(16) _Float16 prob_lds[64 * 64]; // softmax probs

  const int tid  = threadIdx.x;
  const int wv   = tid >> 5;
  const int lane = tid & 31;
  const int nl   = lane & 15;
  const int grp  = (lane >> 4) * 8;

  const int i0 = blockIdx.x * 64;
  const int bh = blockIdx.y;
  const int b  = bh >> 4, h = bh & 15;

  // Q fragments held in registers for the whole block row
  const _Float16* qb = q_ws + ((size_t)bh * S_LEN + i0) * DH;
  const v16h a_q0 = frag_a(qb,      16 * wv, DH);
  const v16h a_q1 = frag_a(qb + 32, 16 * wv, DH);

  float m_r[8], l_r[8];
  v8f acc_o[4];
#pragma unroll
  for (int r = 0; r < 8; ++r) { m_r[r] = -1e30f; l_r[r] = 0.0f; }
#pragma unroll
  for (int n = 0; n < 4; ++n) acc_o[n] = vzero8();

  const int njb = blockIdx.x + 1;          // causal: tiles j0 <= i0
  for (int jb = 0; jb < njb; ++jb) {
    const int j0 = jb * 64;
    const _Float16* kb = k_ws + ((size_t)bh * S_LEN + j0) * DH;
    const _Float16* vb = v_ws + ((size_t)bh * S_LEN + j0) * DH;

    // prefetch next tile's K/V rows into L2 while this tile is processed
    if (jb + 1 < njb) {
      __builtin_prefetch(kb + (size_t)64 * DH + tid * 32, 0, 1);
      __builtin_prefetch(vb + (size_t)64 * DH + tid * 32, 0, 1);
    }

    // ---- K tile: straight f16 copy via async global->LDS (ASYNCcnt) ----
#pragma unroll
    for (int t = 0; t < 4; ++t) {
      const int idx = tid + 128 * t;       // 512 x 16B chunks
      cp_b128_to_lds((const char*)kb + idx * 16, (char*)k_lds + idx * 16);
    }

    // ---- V tile transposed: vectorized b128 reads, b16 scatter into [d][j] ----
#pragma unroll
    for (int t = 0; t < 4; ++t) {
      const int idx = tid + 128 * t;       // 512 chunks of 8 halfs
      const int j = idx >> 3, d0 = (idx & 7) * 8;
      const v8h vv = *(const v8h*)&vb[j * DH + d0];
#pragma unroll
      for (int u = 0; u < 8; ++u) vt_lds[(d0 + u) * 64 + j] = vv[u];
    }

    // ---- Er tile: float4 loads, packed 4xf16 stores ----
    {
      const int base = (S_LEN - 64) + j0 - i0;   // Er row for r = jj-ii+63 == 0
#pragma unroll
      for (int t = 0; t < 16; ++t) {
        const int idx = tid + 128 * t;           // 2048 float4 chunks of 128x64
        const int r = idx >> 4, c4 = (idx & 15) * 4;
        const int g = base + r;
        v4h hv;
        if (g < MAXLEN) {
          hv = cvt4(*(const float4*)&Er[(size_t)g * DH + c4]);
        } else {
          hv[0] = hv[1] = hv[2] = hv[3] = (_Float16)0.0f;
        }
        *(v4h*)&e_lds[idx * 4] = hv;
      }
    }
    async_wait_all();
    __syncthreads();

    // ---- rel GEMM: P = Q_strip (16x64) x E_loc^T (64x128), stage in LDS ----
#pragma unroll
    for (int ct = 0; ct < 8; ++ct) {
      v8f pc = vzero8();
      pc = __builtin_amdgcn_wmma_f32_16x16x32_f16(
          false, a_q0, false, frag_b(e_lds,      ct * 16, DH), (short)0, pc, false, false);
      pc = __builtin_amdgcn_wmma_f32_16x16x32_f16(
          false, a_q1, false, frag_b(e_lds + 32, ct * 16, DH), (short)0, pc, false, false);
#pragma unroll
      for (int r = 0; r < 8; ++r) {
        const int ii = 16 * wv + r + grp;
        p_lds[ii * 128 + ct * 16 + nl] = (_Float16)pc[r];
      }
    }

    // ---- QK^T ----
    v8f sc[4];
#pragma unroll
    for (int c = 0; c < 4; ++c) {
      sc[c] = vzero8();
      sc[c] = __builtin_amdgcn_wmma_f32_16x16x32_f16(
          false, a_q0, false, frag_b(k_lds,      c * 16, DH), (short)0, sc[c], false, false);
      sc[c] = __builtin_amdgcn_wmma_f32_16x16x32_f16(
          false, a_q1, false, frag_b(k_lds + 32, c * 16, DH), (short)0, sc[c], false, false);
    }
    __syncthreads();

    // ---- assemble scores (QK + skewed rel), causal mask, online softmax ----
#pragma unroll
    for (int r = 0; r < 8; ++r) {
      const int ii = 16 * wv + r + grp;
      const int i  = i0 + ii;
#pragma unroll
      for (int c = 0; c < 4; ++c) {
        const int jj = c * 16 + nl;
        const int j  = j0 + jj;
        float s = (sc[c][r] + (float)p_lds[ii * 128 + (jj - ii + 63)]) * 0.125f;
        if (j > i) s = -1e30f;
        sc[c][r] = s;
      }
      float v = fmaxf(fmaxf(sc[0][r], sc[1][r]), fmaxf(sc[2][r], sc[3][r]));
      v = red_max16(v);
      const float mn    = fmaxf(m_r[r], v);
      const float alpha = __expf(m_r[r] - mn);
      m_r[r] = mn;
      float rs = 0.0f;
#pragma unroll
      for (int c = 0; c < 4; ++c) {
        const float p = __expf(sc[c][r] - mn);
        sc[c][r] = p;
        rs += p;
      }
      rs = red_sum16(rs);
      l_r[r] = l_r[r] * alpha + rs;
#pragma unroll
      for (int n = 0; n < 4; ++n) acc_o[n][r] *= alpha;
#pragma unroll
      for (int c = 0; c < 4; ++c)
        prob_lds[ii * 64 + c * 16 + nl] = (_Float16)sc[c][r];
    }
    __syncthreads();

    // ---- P @ V : A = probs (16x64 strip), B = V^T tiles [d][j] ----
    const v16h ap0 = frag_a(prob_lds,      16 * wv, 64);
    const v16h ap1 = frag_a(prob_lds + 32, 16 * wv, 64);
#pragma unroll
    for (int n = 0; n < 4; ++n) {
      acc_o[n] = __builtin_amdgcn_wmma_f32_16x16x32_f16(
          false, ap0, false, frag_b(vt_lds,      n * 16, 64), (short)0, acc_o[n], false, false);
      acc_o[n] = __builtin_amdgcn_wmma_f32_16x16x32_f16(
          false, ap1, false, frag_b(vt_lds + 32, n * 16, 64), (short)0, acc_o[n], false, false);
    }
    __syncthreads();
  }

  // ---- epilogue: out[b, i, h*64+d] = acc / l ----
#pragma unroll
  for (int r = 0; r < 8; ++r) {
    const int ii  = 16 * wv + r + grp;
    const int i   = i0 + ii;
    const float inv = 1.0f / l_r[r];
#pragma unroll
    for (int n = 0; n < 4; ++n) {
      out[((size_t)b * S_LEN + i) * DM + h * DH + n * 16 + nl] = acc_o[n][r] * inv;
    }
  }
}

extern "C" void kernel_launch(void* const* d_in, const int* in_sizes, int n_in,
                              void* d_out, int out_size, void* d_ws, size_t ws_size,
                              hipStream_t stream) {
  const float* query = (const float*)d_in[0];
  const float* key   = (const float*)d_in[1];
  const float* value = (const float*)d_in[2];
  const float* Wq    = (const float*)d_in[3];
  const float* bq    = (const float*)d_in[4];
  const float* Wk    = (const float*)d_in[5];
  const float* bk    = (const float*)d_in[6];
  const float* Wv    = (const float*)d_in[7];
  const float* bv    = (const float*)d_in[8];
  const float* Er    = (const float*)d_in[9];

  const size_t per = (size_t)NB * NH * S_LEN * DH;   // 4,194,304 halfs
  _Float16* q_ws = (_Float16*)d_ws;
  _Float16* k_ws = q_ws + per;
  _Float16* v_ws = k_ws + per;

  dim3 gp(64, 16);   // M=4096 rows / 64, N=1024 / 64
  proj_kernel<<<gp, 128, 0, stream>>>(query, Wq, bq, q_ws);
  proj_kernel<<<gp, 128, 0, stream>>>(key,   Wk, bk, k_ws);
  proj_kernel<<<gp, 128, 0, stream>>>(value, Wv, bv, v_ws);

  dim3 ga(S_LEN / 64, NB * NH);   // (32, 32)
  attn_kernel<<<ga, 128, 0, stream>>>(q_ws, k_ws, v_ws, Er, (float*)d_out);
}
